// ModulatedConv2d_223338299777
// MI455X (gfx1250) — matrix operand order repaired and checked by reading, not compiled, not run on hardware
//
#include <hip/hip_runtime.h>
#include <hip/hip_bf16.h>

typedef __bf16        v16bf __attribute__((ext_vector_type(16)));
typedef float         v8f   __attribute__((ext_vector_type(8)));
typedef unsigned int  v8u   __attribute__((ext_vector_type(8)));

#define C_IN  512
#define C_OUT 512
#define HW    64

// round-to-nearest-even f32 -> bf16 (bit pattern)
__device__ __forceinline__ unsigned short f2bf(float f) {
    unsigned int u = __builtin_bit_cast(unsigned int, f);
    u += 0x7FFFu + ((u >> 16) & 1u);
    return (unsigned short)(u >> 16);
}

// smod[b][c] = (w[b,:] . affine_w[c,:]) + affine_b[c] + 1
__global__ void style_kernel(const float* __restrict__ w,
                             const float* __restrict__ aw,
                             const float* __restrict__ ab,
                             float* __restrict__ smod, int B) {
    int idx = blockIdx.x * blockDim.x + threadIdx.x;
    if (idx >= B * C_IN) return;
    int b = idx / C_IN, c = idx % C_IN;
    const float* wr = w  + b * 512;
    const float* ar = aw + c * 512;
    float acc = 0.f;
    for (int j = 0; j < 512; ++j) acc += wr[j] * ar[j];
    smod[idx] = acc + ab[c] + 1.0f;
}

// wsq[o][c] = sum over 9 taps of weight^2
__global__ void wsq_kernel(const float* __restrict__ weight,
                           float* __restrict__ wsq) {
    int idx = blockIdx.x * blockDim.x + threadIdx.x;  // o*512 + c
    if (idx >= C_OUT * C_IN) return;
    const float* p = weight + (long)idx * 9;
    float s = 0.f;
    #pragma unroll
    for (int t = 0; t < 9; ++t) { float v = p[t]; s += v * v; }
    wsq[idx] = s;
}

// sigma[b][o] = rsqrt( sum_c wsq[o][c] * smod[b][c]^2 + eps )
__global__ void sigma_kernel(const float* __restrict__ wsq,
                             const float* __restrict__ smod,
                             float* __restrict__ sigma, int B) {
    int idx = blockIdx.x * blockDim.x + threadIdx.x;
    if (idx >= B * C_OUT) return;
    int b = idx / C_OUT, o = idx % C_OUT;
    const float* wr = wsq  + o * C_IN;
    const float* sr = smod + b * C_IN;
    float s = 0.f;
    for (int c = 0; c < C_IN; ++c) { float m = sr[c]; s += wr[c] * m * m; }
    sigma[idx] = rsqrtf(s + 1e-8f);
}

// wbf[t][o][c] = bf16(weight[o][c][t])  (c contiguous)
__global__ void wcvt_kernel(const float* __restrict__ weight,
                            unsigned short* __restrict__ wbf) {
    int idx = blockIdx.x * blockDim.x + threadIdx.x;  // (t*512 + o)*512 + c
    if (idx >= 9 * C_OUT * C_IN) return;
    int c = idx % C_IN;
    int o = (idx / C_IN) % C_OUT;
    int t = idx / (C_IN * C_OUT);
    wbf[idx] = f2bf(weight[((long)o * C_IN + c) * 9 + t]);
}

// Implicit-GEMM modulated conv.
// One workgroup = 64 out-ch x (2 rows x 64 px) at rows {h0, h0+1}.
// 8 waves: wave wv -> m_sub = wv&3 (16 o-rows), row_sel = wv>>2 (output row).
// Each wave: 4 accumulators covering its full 64-px row -> A reused 4x per tap.
__global__ __launch_bounds__(256)
void modconv_kernel(const float* __restrict__ xg,
                    const unsigned short* __restrict__ wbf,
                    const float* __restrict__ smod,
                    const float* __restrict__ sigma,
                    const float* __restrict__ bias,
                    float* __restrict__ out) {
    // x tile: 4 rows (h0-1..h0+2) x (64+pad) px x 32 ch, c contiguous
    __shared__ unsigned short Xlds[4][68][32];
    // weight tiles: 9 taps x 64 o x 32 c (as 16 dwords)
    __shared__ unsigned int   Wlds[9][64][16];

    const int tid    = threadIdx.x;
    const int o_base = blockIdx.x * 64;
    const int h0     = blockIdx.y * 2;
    const int b      = blockIdx.z;

    // zero pad columns 0, 65, 66, 67 once (never overwritten afterwards)
    for (int e = tid; e < 4 * 4 * 32; e += 256) {
        int cc  = e & 31;
        int sel = (e >> 5) & 3;
        int dh  = e >> 7;
        int col = (sel == 0) ? 0 : 64 + sel;
        Xlds[dh][col][cc] = 0;
    }

    const int wv      = tid >> 5;
    const int ln      = tid & 31;
    const int hlf     = ln >> 4;      // lane half (0/1)
    const int nl      = ln & 15;      // M index for A, N index for B/C
    const int m_sub   = wv & 3;       // 16-row o subtile
    const int row_sel = wv >> 2;      // which of the 2 output rows

    v8f acc0 = {}, acc1 = {}, acc2 = {}, acc3 = {};

    const unsigned int* wbfU = (const unsigned int*)wbf;

    for (int c0 = 0; c0 < C_IN; c0 += 32) {
        __syncthreads();   // previous chunk's reads done before overwrite

        // ---- stage bf16 weight tiles: 9*64*16 dwords ----
        for (int e = tid; e < 9 * 64 * 16; e += 256) {
            int j  = e & 15;
            int oo = (e >> 4) & 63;
            int t  = e >> 10;
            Wlds[t][oo][j] =
                wbfU[(((t * C_OUT) + o_base + oo) * C_IN + c0) / 2 + j];
        }

        // ---- stage modulated input: 4 rows x 32 ch x 64 px ----
        for (int e = tid; e < 4 * 32 * 64; e += 256) {
            int w_ = e & 63;
            int cc = (e >> 6) & 31;
            int dh = e >> 11;
            int row = h0 + dh - 1;
            int c   = c0 + cc;
            float v = 0.f;
            if (row >= 0 && row < HW) {
                const float* p = &xg[(((long)b * C_IN + c) * HW + row) * HW + w_];
                v = *p * smod[b * C_IN + c];
                if (c + 32 < C_IN)
                    __builtin_prefetch(p + 32 * HW * HW, 0, 0); // next c-chunk
            }
            Xlds[dh][w_ + 1][cc] = f2bf(v);
        }
        __syncthreads();

        // ---- 9 shifted GEMM taps, K = 32, 4 N-subtiles per wave ----
        #pragma unroll
        for (int t = 0; t < 9; ++t) {
            const int kh = t / 3, kw = t % 3;
            const int xrow = row_sel + kh;   // 0..3

            // A fragment: 16x32 bf16 (V0-3 = K0-7|8-15, V4-7 = K16-23|24-31)
            v8u au;
            #pragma unroll
            for (int i = 0; i < 8; ++i)
                au[i] = Wlds[t][m_sub * 16 + nl][(i < 4 ? i : i + 4) + 4 * hlf];
            v16bf A = __builtin_bit_cast(v16bf, au);

            // B fragments: 32x16 bf16 (V0-3 = K0-7|8-15, V4-7 = K16-23|24-31)
            const unsigned int* xr0 = (const unsigned int*)&Xlds[xrow][ 0 + nl + kw][0];
            const unsigned int* xr1 = (const unsigned int*)&Xlds[xrow][16 + nl + kw][0];
            const unsigned int* xr2 = (const unsigned int*)&Xlds[xrow][32 + nl + kw][0];
            const unsigned int* xr3 = (const unsigned int*)&Xlds[xrow][48 + nl + kw][0];
            v8u bu0, bu1, bu2, bu3;
            #pragma unroll
            for (int i = 0; i < 8; ++i) {
                int d = (i < 4 ? i : i + 4) + 4 * hlf;
                bu0[i] = xr0[d];
                bu1[i] = xr1[d];
                bu2[i] = xr2[d];
                bu3[i] = xr3[d];
            }
            v16bf B0 = __builtin_bit_cast(v16bf, bu0);
            v16bf B1 = __builtin_bit_cast(v16bf, bu1);
            v16bf B2 = __builtin_bit_cast(v16bf, bu2);
            v16bf B3 = __builtin_bit_cast(v16bf, bu3);

            acc0 = __builtin_amdgcn_wmma_f32_16x16x32_bf16(
                false, A, false, B0, (short)0, acc0, false, false);
            acc1 = __builtin_amdgcn_wmma_f32_16x16x32_bf16(
                false, A, false, B1, (short)0, acc1, false, false);
            acc2 = __builtin_amdgcn_wmma_f32_16x16x32_bf16(
                false, A, false, B2, (short)0, acc2, false, false);
            acc3 = __builtin_amdgcn_wmma_f32_16x16x32_bf16(
                false, A, false, B3, (short)0, acc3, false, false);
        }
    }

    // ---- demodulate + bias + store (C layout: VGPR r -> M = r + 8*half) ----
    const int h = h0 + row_sel;
    #pragma unroll
    for (int r = 0; r < 8; ++r) {
        int o = o_base + m_sub * 16 + r + 8 * hlf;
        float sg = sigma[b * C_OUT + o];
        float bo = bias[o];
        long base = (((long)b * C_OUT + o) * HW + h) * HW + nl;
        out[base]      = acc0[r] * sg + bo;
        out[base + 16] = acc1[r] * sg + bo;
        out[base + 32] = acc2[r] * sg + bo;
        out[base + 48] = acc3[r] * sg + bo;
    }
}

extern "C" void kernel_launch(void* const* d_in, const int* in_sizes, int n_in,
                              void* d_out, int out_size, void* d_ws, size_t ws_size,
                              hipStream_t stream) {
    const float* x    = (const float*)d_in[0];
    const float* w    = (const float*)d_in[1];
    const float* wt   = (const float*)d_in[2];
    const float* bias = (const float*)d_in[3];
    const float* aw   = (const float*)d_in[4];
    const float* ab   = (const float*)d_in[5];
    float* out = (float*)d_out;

    const int B = in_sizes[0] / (C_IN * HW * HW);

    // workspace layout
    char* ws = (char*)d_ws;
    float* smod  = (float*)(ws);                          // B*512 f32 (<=32KB)
    float* sigma = (float*)(ws + 32768);                  // B*512 f32 (<=32KB)
    float* wsq   = (float*)(ws + 65536);                  // 512*512 f32 = 1MB
    unsigned short* wbf =
        (unsigned short*)(ws + 65536 + C_OUT * C_IN * 4); // 9*512*512 bf16 = 4.5MB

    style_kernel<<<(B * C_IN + 255) / 256, 256, 0, stream>>>(w, aw, ab, smod, B);
    wsq_kernel<<<(C_OUT * C_IN + 255) / 256, 256, 0, stream>>>(wt, wsq);
    sigma_kernel<<<(B * C_OUT + 255) / 256, 256, 0, stream>>>(wsq, smod, sigma, B);
    wcvt_kernel<<<(9 * C_OUT * C_IN + 255) / 256, 256, 0, stream>>>(wt, wbf);

    dim3 grid(C_OUT / 64, HW / 2, B);
    modconv_kernel<<<grid, 256, 0, stream>>>(x, wbf, smod, sigma, bias, out);
}
